// Net2_78941498900895
// MI455X (gfx1250) — matrix operand order repaired and checked by reading, not compiled, not run on hardware
//
#include <hip/hip_runtime.h>
#include <hip/hip_bf16.h>

typedef _Float16 h16;
typedef __attribute__((ext_vector_type(16))) _Float16 v16h;
typedef __attribute__((ext_vector_type(8)))  _Float16 v8h;
typedef __attribute__((ext_vector_type(8)))  float    v8f;

#define N_STK 10000
#define KF    512
#define NIND  30
#define MT    625          // N_STK / 16

// ---------------------------------------------------------------------------
// zero a span of 32-bit words (stats region must be zeroed every call)
__global__ void zero_kernel(float* p, int n) {
    int i = blockIdx.x * blockDim.x + threadIdx.x;
    if (i < n) p[i] = 0.0f;
}

// ---------------------------------------------------------------------------
// rf16 = (h16) clip(raw-0.5, -1, 1); ind_id from one-hot; industry counts
__global__ void prep_kernel(const float* __restrict__ raw,
                            const float* __restrict__ dummy,
                            h16* __restrict__ rf16,
                            int* __restrict__ ind_id, int* __restrict__ cnt) {
    int row = blockIdx.x;
    const float* src = raw + (size_t)row * KF;
    h16* dst = rf16 + (size_t)row * KF;
    for (int k = threadIdx.x; k < KF; k += blockDim.x) {
        float v = src[k] - 0.5f;
        v = fminf(1.0f, fmaxf(-1.0f, v));
        dst[k] = (h16)v;
    }
    if (threadIdx.x == 0) {
        const float* d = dummy + (size_t)row * NIND;
        int g = 0;
        for (int j = 0; j < NIND; j++) if (d[j] > 0.5f) g = j;
        ind_id[row] = g;
        atomicAdd(&cnt[g], 1);
    }
}

// ---------------------------------------------------------------------------
// per-industry column sums, accumulated in LDS (thread t owns column t -> no races)
__global__ void isum_kernel(const h16* __restrict__ rf16,
                            const int* __restrict__ ind_id,
                            float* __restrict__ S) {
    __shared__ float lsum[NIND * KF];           // 61440 B
    int t = threadIdx.x;                        // blockDim == 512
    for (int i = t; i < NIND * KF; i += blockDim.x) lsum[i] = 0.0f;
    __syncthreads();
    int per = (N_STK + gridDim.x - 1) / gridDim.x;
    int r0 = blockIdx.x * per;
    int r1 = r0 + per; if (r1 > N_STK) r1 = N_STK;
    for (int r = r0; r < r1; r++) {
        int g = ind_id[r];
        lsum[g * KF + t] += (float)rf16[(size_t)r * KF + t];
    }
    __syncthreads();
    for (int i = t; i < NIND * KF; i += blockDim.x) atomicAdd(&S[i], lsum[i]);
}

// ---------------------------------------------------------------------------
// rfg16[i,k] = rf - (S_g - rf) / (cnt_g - 1 + 1e-8)
__global__ void gf_kernel(const h16* __restrict__ rf16,
                          const int* __restrict__ ind_id,
                          const float* __restrict__ S,
                          const int* __restrict__ cnt,
                          h16* __restrict__ rfg16) {
    int row = blockIdx.x;
    int g = ind_id[row];
    float inv = 1.0f / ((float)(cnt[g] - 1) + 1e-8f);
    for (int k = threadIdx.x; k < KF; k += blockDim.x) {
        float r = (float)rf16[(size_t)row * KF + k];
        float gfv = (S[g * KF + k] - r) * inv;
        rfg16[(size_t)row * KF + k] = (h16)(r - gfv);
    }
}

// ---------------------------------------------------------------------------
__global__ void wconv_kernel(const float* __restrict__ src, h16* __restrict__ dst, int n) {
    for (int i = blockIdx.x * blockDim.x + threadIdx.x; i < n; i += gridDim.x * blockDim.x)
        dst[i] = (h16)src[i];
}

// ---------------------------------------------------------------------------
// WMMA GEMM: out[m,n] = epilogue( sum_k A[m,k] * W[n,k] + bias[n] )
// One wave (32 threads) computes a 16x64 output strip, 4 independent accumulators.
// A is split at splitK between A0 and A1 (both with row stride lda) so layer 1
// reads concat(feat, rf-gf) without materializing the concat.
// EPI==0: cross  -> d = rf*(acc+b) + prev     (rf/prev f16 buffers, stride Nout)
// EPI==1: silu   -> d = x * sigmoid(x)
template <int EPI>
__global__ __launch_bounds__(32) void wmma_gemm(
        const h16* __restrict__ A0, const h16* __restrict__ A1, int splitK, int lda,
        const h16* __restrict__ Bw, const float* __restrict__ bias,
        int Nout, int Kin,
        h16* __restrict__ out16,
        const h16* __restrict__ rf16, const h16* __restrict__ prev16) {
    int mtile = blockIdx.x;
    int nbase = blockIdx.y * 64;
    int l  = threadIdx.x;
    int lo = l & 15;
    int hi = (l >> 4) & 1;
    int rowA = mtile * 16 + lo;

    v8f acc[4] = {};

    int nk = Kin >> 5;
    for (int kc = 0; kc < nk; kc++) {
        int kbase = kc * 32;
        const h16* Ap; int koff;
        if (kbase < splitK) { Ap = A0; koff = kbase; }
        else                { Ap = A1; koff = kbase - splitK; }

        // A 16x32: lane holds two contiguous 8-half chunks (ISA A-matrix layout)
        const h16* pa = Ap + (size_t)rowA * lda + koff + hi * 8;
        v8h alo = *(const v8h*)pa;
        v8h ahi = *(const v8h*)(pa + 16);
        v16h a;
        #pragma unroll
        for (int i = 0; i < 8; i++) { a[i] = alo[i]; a[i + 8] = ahi[i]; }

        const h16* pbBase = Bw + (size_t)(nbase + lo) * Kin + kbase + hi * 8;
        #pragma unroll
        for (int t = 0; t < 4; t++) {
            const h16* pb = pbBase + (size_t)t * 16 * Kin;
            v8h b0 = *(const v8h*)pb;
            v8h b1 = *(const v8h*)(pb + 16);
            v16h b;
            #pragma unroll
            for (int i = 0; i < 8; i++) { b[i] = b0[i]; b[i + 8] = b1[i]; }
            acc[t] = __builtin_amdgcn_wmma_f32_16x16x32_f16(
                false, a, false, b, (short)0, acc[t], false, false);
        }
    }

    // epilogue: C/D layout -> lane(l), vgpr(r): M = r + 8*hi, N = lo
    #pragma unroll
    for (int t = 0; t < 4; t++) {
        int n = nbase + t * 16 + lo;
        float bn = bias[n];
        #pragma unroll
        for (int r = 0; r < 8; r++) {
            int m = mtile * 16 + (hi ? r + 8 : r);
            float v = acc[t][r] + bn;
            float o;
            if (EPI == 0) {
                float rfv = (float)rf16[(size_t)m * Nout + n];
                float pv  = (float)prev16[(size_t)m * Nout + n];
                o = rfv * v + pv;
            } else {
                o = v / (1.0f + __expf(-v));      // SiLU
            }
            out16[(size_t)m * Nout + n] = (h16)o;
        }
    }
}

// ---------------------------------------------------------------------------
__global__ void head_kernel(const h16* __restrict__ h3, const float* __restrict__ W4,
                            const float* __restrict__ b4, float* __restrict__ out_raw) {
    int i = blockIdx.x * blockDim.x + threadIdx.x;
    if (i >= N_STK) return;
    float acc = b4[0];
    const h16* p = h3 + (size_t)i * 256;
    #pragma unroll 8
    for (int k = 0; k < 256; k++) acc += (float)p[k] * W4[k];
    out_raw[i] = acc;
}

__global__ void reduce_kernel(const float* __restrict__ x, int n, float* __restrict__ red) {
    __shared__ float s1[256], s2[256];
    float a = 0.f, b = 0.f;
    for (int i = blockIdx.x * blockDim.x + threadIdx.x; i < n; i += gridDim.x * blockDim.x) {
        float v = x[i]; a += v; b += v * v;
    }
    s1[threadIdx.x] = a; s2[threadIdx.x] = b;
    __syncthreads();
    for (int off = 128; off > 0; off >>= 1) {
        if (threadIdx.x < off) {
            s1[threadIdx.x] += s1[threadIdx.x + off];
            s2[threadIdx.x] += s2[threadIdx.x + off];
        }
        __syncthreads();
    }
    if (threadIdx.x == 0) { atomicAdd(&red[0], s1[0]); atomicAdd(&red[1], s2[0]); }
}

__global__ void bn_kernel(const float* __restrict__ x, int n, const float* __restrict__ red,
                          const float* __restrict__ gamma, const float* __restrict__ beta,
                          float* __restrict__ out) {
    int i = blockIdx.x * blockDim.x + threadIdx.x;
    if (i >= n) return;
    float mu  = red[0] / (float)n;
    float var = red[1] / (float)n - mu * mu;
    float scale = rsqrtf(var + 1e-5f) * gamma[0];
    out[i] = (x[i] - mu) * scale + beta[0];
}

// ---------------------------------------------------------------------------
extern "C" void kernel_launch(void* const* d_in, const int* in_sizes, int n_in,
                              void* d_out, int out_size, void* d_ws, size_t ws_size,
                              hipStream_t stream) {
    const float* raw     = (const float*)d_in[0];   // [10000,512]
    const float* dummy   = (const float*)d_in[1];   // [10000,30]
    const float* cross_W = (const float*)d_in[2];   // [3,512,512]
    const float* cross_b = (const float*)d_in[3];   // [3,512]
    const float* W1 = (const float*)d_in[4];        // [1024,1024]
    const float* b1 = (const float*)d_in[5];
    const float* W2 = (const float*)d_in[6];        // [512,1024]
    const float* b2 = (const float*)d_in[7];
    const float* W3 = (const float*)d_in[8];        // [256,512]
    const float* b3 = (const float*)d_in[9];
    const float* W4 = (const float*)d_in[10];       // [1,256]
    const float* b4 = (const float*)d_in[11];
    const float* gamma = (const float*)d_in[12];
    const float* beta  = (const float*)d_in[13];
    float* out = (float*)d_out;

    // ---- workspace bump allocator (256B aligned) ----
    size_t off = 0;
    char* base = (char*)d_ws;
    auto alloc = [&](size_t bytes) -> void* {
        void* p = base + off;
        off = (off + bytes + 255) & ~(size_t)255;
        return p;
    };
    h16* rf16  = (h16*)alloc((size_t)N_STK * KF * 2);
    h16* rfg16 = (h16*)alloc((size_t)N_STK * KF * 2);
    h16* featA = (h16*)alloc((size_t)N_STK * KF * 2);
    h16* featB = (h16*)alloc((size_t)N_STK * KF * 2);
    h16* h1    = (h16*)alloc((size_t)N_STK * 1024 * 2);
    h16* h2    = (h16*)alloc((size_t)N_STK * 512 * 2);
    h16* h3    = (h16*)alloc((size_t)N_STK * 256 * 2);
    h16* cw    = (h16*)alloc((size_t)3 * 512 * 512 * 2);
    h16* W1h   = (h16*)alloc((size_t)1024 * 1024 * 2);
    h16* W2h   = (h16*)alloc((size_t)512 * 1024 * 2);
    h16* W3h   = (h16*)alloc((size_t)256 * 512 * 2);
    float* S   = (float*)alloc((size_t)(NIND * KF + 32 + 16) * 4);   // S | cnt | red
    int*   cnt = (int*)(S + NIND * KF);
    float* red = (float*)(cnt + 32);
    int*   ind_id  = (int*)alloc((size_t)N_STK * 4);
    float* out_raw = (float*)alloc((size_t)N_STK * 4);
    (void)ws_size; (void)in_sizes; (void)n_in; (void)out_size;

    const int statWords = NIND * KF + 32 + 16;

    // ---- pipeline ----
    zero_kernel<<<(statWords + 255) / 256, 256, 0, stream>>>(S, statWords);
    prep_kernel<<<N_STK, 256, 0, stream>>>(raw, dummy, rf16, ind_id, cnt);
    isum_kernel<<<64, 512, 0, stream>>>(rf16, ind_id, S);
    gf_kernel<<<N_STK, 256, 0, stream>>>(rf16, ind_id, S, cnt, rfg16);

    wconv_kernel<<<512, 256, 0, stream>>>(cross_W, cw, 3 * 512 * 512);
    wconv_kernel<<<512, 256, 0, stream>>>(W1, W1h, 1024 * 1024);
    wconv_kernel<<<512, 256, 0, stream>>>(W2, W2h, 512 * 1024);
    wconv_kernel<<<256, 256, 0, stream>>>(W3, W3h, 256 * 512);

    // cross layers: feat = rf * (feat @ Wc^T + bc) + feat
    wmma_gemm<0><<<dim3(MT, 8), 32, 0, stream>>>(rf16, rf16, 512, 512,
        cw,              cross_b,        512, 512, featA, rf16, rf16);
    wmma_gemm<0><<<dim3(MT, 8), 32, 0, stream>>>(featA, featA, 512, 512,
        cw + 262144,     cross_b + 512,  512, 512, featB, rf16, featA);
    wmma_gemm<0><<<dim3(MT, 8), 32, 0, stream>>>(featB, featB, 512, 512,
        cw + 524288,     cross_b + 1024, 512, 512, featA, rf16, featB);

    // MLP: concat(feat, rf-gf) -> 1024 -> 512 -> 256, SiLU
    wmma_gemm<1><<<dim3(MT, 16), 32, 0, stream>>>(featA, rfg16, 512, 512,
        W1h, b1, 1024, 1024, h1, featA, featA);
    wmma_gemm<1><<<dim3(MT, 8), 32, 0, stream>>>(h1, h1, 1024, 1024,
        W2h, b2, 512, 1024, h2, h1, h1);
    wmma_gemm<1><<<dim3(MT, 4), 32, 0, stream>>>(h2, h2, 512, 512,
        W3h, b3, 256, 512, h3, h2, h2);

    head_kernel<<<(N_STK + 255) / 256, 256, 0, stream>>>(h3, W4, b4, out_raw);
    reduce_kernel<<<40, 256, 0, stream>>>(out_raw, N_STK, red);
    bn_kernel<<<(N_STK + 255) / 256, 256, 0, stream>>>(out_raw, N_STK, red, gamma, beta, out);
}
